// FeatureExtractor_76845554860761
// MI455X (gfx1250) — matrix operand order repaired and checked by reading, not compiled, not run on hardware
//
#include <hip/hip_runtime.h>

typedef __attribute__((ext_vector_type(16))) _Float16 v16h;
typedef __attribute__((ext_vector_type(8)))  _Float16 v8h;
typedef __attribute__((ext_vector_type(8)))  float    v8f;
typedef __attribute__((ext_vector_type(4)))  float    v4f;

namespace {
constexpr int kT = 256, kS = 2048, kC = 128, kHid = 32, kG = 64, kL = 64, kL3 = 192;
constexpr int kSTile   = 32;    // stocks per block
constexpr int kThreads = 128;   // 4 waves (wave32)
// LDS row strides (elements); all rows 16-byte aligned
constexpr int LDXF = 132;                      // fp32 x tile stride (floats)
constexpr int LDW1 = 128, LDW2 = 32, LDWG = 64, LDH1 = 40, LDH2 = 72, LDHH = 72;
}

__device__ __forceinline__ v8f wmma_f16(v16h a, v16h b, v8f c) {
  // D = A(16x32 f16) * B(32x16 f16) + C(16x16 f32)
  return __builtin_amdgcn_wmma_f32_16x16x32_f16(false, a, false, b, (short)0, c, false, false);
}

// A fragment (16x32 f16) from f16 LDS, row-major, stride ld (halves).
// lane l: row = l&15; K = 8*(l>>4)+{0..7} and 16+8*(l>>4)+{0..7}
__device__ __forceinline__ v16h ldsA(const _Float16* p0, int ld, int lane) {
  const int r  = lane & 15;
  const int hi = lane >> 4;
  const _Float16* p = p0 + r * ld + hi * 8;
  v8h lo = *(const v8h*)(p);
  v8h hq = *(const v8h*)(p + 16);
  v16h o;
#pragma unroll
  for (int i = 0; i < 8; ++i) { o[i] = lo[i]; o[i + 8] = hq[i]; }
  return o;
}

// A fragment from fp32 LDS with on-the-fly f32->f16 conversion.
__device__ __forceinline__ v16h ldsA_f32(const float* p0, int ld, int lane) {
  const int r  = lane & 15;
  const int hi = lane >> 4;
  const float* p = p0 + r * ld + hi * 8;
  v4f a0 = *(const v4f*)(p);
  v4f a1 = *(const v4f*)(p + 4);
  v4f b0 = *(const v4f*)(p + 16);
  v4f b1 = *(const v4f*)(p + 20);
  v16h o;
#pragma unroll
  for (int i = 0; i < 4; ++i) {
    o[i]      = (_Float16)a0[i];
    o[i + 4]  = (_Float16)a1[i];
    o[i + 8]  = (_Float16)b0[i];
    o[i + 12] = (_Float16)b1[i];
  }
  return o;
}

// B fragment (32x16 f16) from LDS stored n-major / k-contiguous, stride ld.
// lane l: n = l&15; K = 8*(l>>4)+{0..7} and 16+8*(l>>4)+{0..7}
__device__ __forceinline__ v16h ldsB(const _Float16* p0, int ld, int lane) {
  const int n  = lane & 15;
  const int hi = lane >> 4;
  const _Float16* p = p0 + n * ld + hi * 8;
  v8h a = *(const v8h*)(p);
  v8h b = *(const v8h*)(p + 16);
  v16h o;
#pragma unroll
  for (int i = 0; i < 8; ++i) { o[i] = a[i]; o[i + 8] = b[i]; }
  return o;
}

// Async copy of one (kSTile x kC) fp32 x-tile into LDS (ping-pong buffer).
// Each of 128 threads issues 8 x 16-byte GLOBAL_LOAD_ASYNC_TO_LDS_B128.
__device__ __forceinline__ void async_load_tile(const float* gbase, float* ldsBase, int tid) {
  unsigned lds0 = (unsigned)(unsigned long long)(void*)ldsBase;
#pragma unroll
  for (int j = 0; j < 8; ++j) {
    int i = tid + j * kThreads;        // float4 chunk index, 0..1023
    int r = i >> 5;                    // 32 chunks per row
    int c = (i & 31) << 2;             // float column
    unsigned lds = lds0 + (unsigned)(r * LDXF + c) * 4u;
    unsigned long long ga = (unsigned long long)(gbase + (size_t)i * 4);
    asm volatile("global_load_async_to_lds_b128 %0, %1, off"
                 :: "v"(lds), "v"(ga) : "memory");
  }
}

__device__ __forceinline__ void wait_async0() {
  asm volatile("s_wait_asynccnt 0x0" ::: "memory");
}

__global__ __launch_bounds__(kThreads, 1)
void gru_fused_kernel(const float* __restrict__ x,
                      const float* __restrict__ W1,  const float* __restrict__ b1,
                      const float* __restrict__ W2,  const float* __restrict__ b2,
                      const float* __restrict__ Wih, const float* __restrict__ Whh,
                      const float* __restrict__ bih, const float* __restrict__ bhh,
                      float* __restrict__ out) {
  __shared__ float    sXf[2][kSTile * LDXF]; // fp32 x tile, ping-pong (async target)
  __shared__ _Float16 sW1[kHid * LDW1];      // [n][k] = W1[k][n]
  __shared__ _Float16 sW2[kG * LDW2];        // [n][k] = W2[k][n]
  __shared__ _Float16 sWih[kL3 * LDWG];      // [j][g] = W_ih (row-major already k-contig)
  __shared__ _Float16 sWhh[kL3 * LDWG];
  __shared__ _Float16 sH1[kSTile * LDH1];
  __shared__ _Float16 sH2[kSTile * LDH2];
  __shared__ _Float16 sHh[2][kSTile * LDHH]; // hidden, f16 ping-pong (WMMA A operand)
  __shared__ float    sHf[2][kSTile * kL];   // hidden, fp32 master ping-pong
  __shared__ float    sB1[kHid];
  __shared__ float    sB2[kG];
  __shared__ float    sBih[kL3];
  __shared__ float    sBhh[kL3];

  const int tid   = threadIdx.x;
  const int wave  = tid >> 5;
  const int lane  = tid & 31;
  const int sBase = blockIdx.x * kSTile;
  const float* xBlk = x + (size_t)sBase * kC;   // + t*kS*kC per step

  // Kick off async fetch of tile t=0 while we stage weights.
  async_load_tile(xBlk, sXf[0], tid);

  // ---- one-time staging: weights f32->f16 (B layout), biases, h0 = 0 ----
  for (int i = tid; i < kC * kHid; i += kThreads) {
    int k = i >> 5, n = i & 31;
    sW1[n * LDW1 + k] = (_Float16)W1[k * kHid + n];
  }
  for (int i = tid; i < kHid * kG; i += kThreads) {
    int k = i >> 6, n = i & 63;
    sW2[n * LDW2 + k] = (_Float16)W2[k * kG + n];
  }
  for (int i = tid; i < kL3 * kG; i += kThreads) {
    sWih[i] = (_Float16)Wih[i];
    sWhh[i] = (_Float16)Whh[i];
  }
  for (int i = tid; i < kHid; i += kThreads) sB1[i] = b1[i];
  for (int i = tid; i < kG;   i += kThreads) sB2[i] = b2[i];
  for (int i = tid; i < kL3;  i += kThreads) { sBih[i] = bih[i]; sBhh[i] = bhh[i]; }
  for (int i = tid; i < kSTile * kL; i += kThreads) { sHf[0][i] = 0.f; sHf[1][i] = 0.f; }
  for (int i = tid; i < kSTile * LDHH; i += kThreads) {
    sHh[0][i] = (_Float16)0.f; sHh[1][i] = (_Float16)0.f;
  }
  __syncthreads();

  const int mt = wave >> 1;  // row tile (16 rows) handled by this wave

  for (int t = 0; t < kT; ++t) {
    const int p = t & 1;

    // ---- retire this step's async x tile, then prefetch the next one ----
    wait_async0();
    __syncthreads();
    if (t + 1 < kT) {
      async_load_tile(xBlk + (size_t)(t + 1) * kS * kC, sXf[1 - p], tid);
    }
    const float* sX = sXf[p];

    // ---- stage 2: h1 = lrelu(x @ W1 + b1), 32x32 ----
    {
      int nt = wave & 1;
      v8f acc = {};
#pragma unroll
      for (int kc = 0; kc < 4; ++kc) {
        v16h a = ldsA_f32(&sX[mt * 16 * LDXF + kc * 32], LDXF, lane);
        v16h b = ldsB(&sW1[nt * 16 * LDW1 + kc * 32], LDW1, lane);
        acc = wmma_f16(a, b, acc);
      }
      int col = nt * 16 + (lane & 15);
      int rb  = mt * 16 + ((lane >> 4) << 3);
#pragma unroll
      for (int i = 0; i < 8; ++i) {
        float v = acc[i] + sB1[col];
        v = v >= 0.f ? v : 0.01f * v;
        sH1[(rb + i) * LDH1 + col] = (_Float16)v;
      }
    }
    __syncthreads();

    // ---- stage 3: h2 = lrelu(h1 @ W2 + b2), 32x64 ----
    {
      v16h a = ldsA(&sH1[mt * 16 * LDH1], LDH1, lane);
#pragma unroll
      for (int nti = 0; nti < 2; ++nti) {
        int nt = (wave & 1) + 2 * nti;
        v16h b = ldsB(&sW2[nt * 16 * LDW2], LDW2, lane);
        v8f acc = {};
        acc = wmma_f16(a, b, acc);
        int col = nt * 16 + (lane & 15);
        int rb  = mt * 16 + ((lane >> 4) << 3);
#pragma unroll
        for (int i = 0; i < 8; ++i) {
          float v = acc[i] + sB2[col];
          v = v >= 0.f ? v : 0.01f * v;
          sH2[(rb + i) * LDH2 + col] = (_Float16)v;
        }
      }
    }
    __syncthreads();

    // ---- stage 4: gx = h2 @ Wih^T, gh = h @ Whh^T (registers), gates, h update ----
    {
      const float*    hIn   = sHf[p];
      float*          hOut  = sHf[1 - p];
      const _Float16* hhIn  = sHh[p];
      _Float16*       hhOut = sHh[1 - p];
#pragma unroll
      for (int cti = 0; cti < 2; ++cti) {
        int ct = (wave & 1) + 2 * cti;        // 16-col tile within L
        v8f gx0 = {}, gx1 = {}, gx2 = {}, gh0 = {}, gh1 = {}, gh2 = {};
#pragma unroll
        for (int kc = 0; kc < 2; ++kc) {
          v16h a2 = ldsA(&sH2[mt * 16 * LDH2 + kc * 32], LDH2, lane);
          v16h ah = ldsA(&hhIn[mt * 16 * LDHH + kc * 32], LDHH, lane);
          int jb = ct * 16;
          v16h bx0 = ldsB(&sWih[(jb      ) * LDWG + kc * 32], LDWG, lane);
          v16h bx1 = ldsB(&sWih[(jb +  64) * LDWG + kc * 32], LDWG, lane);
          v16h bx2 = ldsB(&sWih[(jb + 128) * LDWG + kc * 32], LDWG, lane);
          v16h bh0 = ldsB(&sWhh[(jb      ) * LDWG + kc * 32], LDWG, lane);
          v16h bh1 = ldsB(&sWhh[(jb +  64) * LDWG + kc * 32], LDWG, lane);
          v16h bh2 = ldsB(&sWhh[(jb + 128) * LDWG + kc * 32], LDWG, lane);
          gx0 = wmma_f16(a2, bx0, gx0);
          gx1 = wmma_f16(a2, bx1, gx1);
          gx2 = wmma_f16(a2, bx2, gx2);
          gh0 = wmma_f16(ah, bh0, gh0);
          gh1 = wmma_f16(ah, bh1, gh1);
          gh2 = wmma_f16(ah, bh2, gh2);
        }
        int col = ct * 16 + (lane & 15);
        int rb  = mt * 16 + ((lane >> 4) << 3);
#pragma unroll
        for (int i = 0; i < 8; ++i) {
          int row = rb + i;
          float grv = gx0[i] + sBih[col]      + gh0[i] + sBhh[col];
          float gzv = gx1[i] + sBih[kL + col] + gh1[i] + sBhh[kL + col];
          float r = 1.f / (1.f + __expf(-grv));
          float z = 1.f / (1.f + __expf(-gzv));
          float n = tanhf(gx2[i] + sBih[2 * kL + col] + r * (gh2[i] + sBhh[2 * kL + col]));
          float hNew = (1.f - z) * n + z * hIn[row * kL + col];
          hOut[row * kL + col]    = hNew;
          hhOut[row * LDHH + col] = (_Float16)hNew;
        }
      }
    }
    __syncthreads();
  }

  // T = 256 (even): final state landed in ping-pong buffer 0
  for (int i = tid; i < kSTile * kL; i += kThreads) {
    int r = i >> 6, c = i & 63;
    out[(size_t)(sBase + r) * kL + c] = sHf[0][r * kL + c];
  }
}

extern "C" void kernel_launch(void* const* d_in, const int* in_sizes, int n_in,
                              void* d_out, int out_size, void* d_ws, size_t ws_size,
                              hipStream_t stream) {
  (void)in_sizes; (void)n_in; (void)out_size; (void)d_ws; (void)ws_size;
  const float* x   = (const float*)d_in[0];
  const float* W1  = (const float*)d_in[1];
  const float* b1  = (const float*)d_in[2];
  const float* W2  = (const float*)d_in[3];
  const float* b2  = (const float*)d_in[4];
  const float* Wih = (const float*)d_in[5];
  const float* Whh = (const float*)d_in[6];
  const float* bih = (const float*)d_in[7];
  const float* bhh = (const float*)d_in[8];
  hipLaunchKernelGGL(gru_fused_kernel, dim3(kS / kSTile), dim3(kThreads), 0, stream,
                     x, W1, b1, W2, b2, Wih, Whh, bih, bhh, (float*)d_out);
}
